// MultiScaleSpatialAttention_39470749450437
// MI455X (gfx1250) — matrix operand order repaired and checked by reading, not compiled, not run on hardware
//
#include <hip/hip_runtime.h>
#include <hip/hip_bf16.h>
#include <math.h>

typedef _Float16 f16;
typedef __attribute__((ext_vector_type(16))) _Float16 v16h;
typedef __attribute__((ext_vector_type(8)))  float    v8f;

#define B_   4
#define C_   128
#define H_   64
#define W_   64
#define I_   32      // inter = C/4
#define G_   8       // groups
#define CPG  16      // channels per group
#define EPS_ 1e-5f
#define S_   3

// gfx1250 async global->LDS path (ASYNCcnt-tracked), if the toolchain declares it.
#if defined(__has_builtin)
#if __has_builtin(__builtin_amdgcn_global_load_async_to_lds_b128) && \
    __has_builtin(__builtin_amdgcn_s_wait_asynccnt)
#define ASYNC_LDS 1
#endif
#endif

// ---------------------------------------------------------------------------
// WMMA fragment loaders, following CDNA5 ISA 7.12.2 wave32 layouts exactly.
// ---------------------------------------------------------------------------

// A fragment: 16x32 f16 (MxK). Row-major source, leading dim ld.
__device__ __forceinline__ v16h load_a_rowmajor(const f16* __restrict__ src, int ld) {
  const int lane = threadIdx.x & 31;
  const int row  = lane & 15;
  const int kh   = (lane >> 4) << 3;   // 0 or 8
  v16h a;
#pragma unroll
  for (int j = 0; j < 8; ++j) {
    const int k = (j < 4) ? (2 * j + kh) : (16 + 2 * (j - 4) + kh);
    a[2 * j]     = src[row * ld + k];
    a[2 * j + 1] = src[row * ld + k + 1];
  }
  return a;
}

// B fragment: 32x16 f16 (KxN) where B[k][n] = src[n*ld + k]  (B = src^T).
__device__ __forceinline__ v16h load_b_transposed(const f16* __restrict__ src, int ld) {
  const int lane = threadIdx.x & 31;
  const int n    = lane & 15;
  const int kh   = (lane >> 4) << 3;   // 0 or 8
  v16h b;
#pragma unroll
  for (int j = 0; j < 8; ++j) {
    const int k = (j < 4) ? (2 * j + kh) : (16 + 2 * (j - 4) + kh);
    b[2 * j]     = src[n * ld + k];
    b[2 * j + 1] = src[n * ld + k + 1];
  }
  return b;
}

__device__ __forceinline__ v8f wmma16(v16h a, v16h b, v8f c) {
  return __builtin_amdgcn_wmma_f32_16x16x32_f16(false, a, false, b, (short)0, c,
                                                false, false);
}

// 16-byte global -> LDS staging helper
#ifdef ASYNC_LDS
// Builtin signature (from the toolchain's own diagnostic): first param is
// 'int __vector_size__(16) __device__ *' == AS(1) int4*; LDS side is AS(3).
typedef int v4i_ __attribute__((vector_size(16)));
__device__ __forceinline__ void stage16(const f16* __restrict__ g, f16* __restrict__ l) {
  typedef __attribute__((address_space(1))) v4i_ gvec;
  typedef __attribute__((address_space(3))) v4i_ lvec;
  __builtin_amdgcn_global_load_async_to_lds_b128((gvec*)g, (lvec*)l, 0, 0);
}
#else
__device__ __forceinline__ void stage16(const f16* __restrict__ g, f16* __restrict__ l) {
  *(uint4*)(void*)l = *(const uint4*)(const void*)g;
}
#endif

// ---------------------------------------------------------------------------
// 1. f32 -> f16 weight conversion
// ---------------------------------------------------------------------------
__global__ void k_f32_to_f16(const float* __restrict__ in, f16* __restrict__ out, int n) {
  int i = blockIdx.x * blockDim.x + threadIdx.x;
  if (i < n) out[i] = (f16)in[i];
}

// ---------------------------------------------------------------------------
// 2. 2x2 average pool (applied twice: 64->32->16; mean-of-means == 4x4 mean)
// ---------------------------------------------------------------------------
__global__ void k_down2(const float* __restrict__ in, float* __restrict__ out, int hin, int win) {
  const int ho = hin >> 1, wo = win >> 1;
  const int total = B_ * C_ * ho * wo;
  int i = blockIdx.x * blockDim.x + threadIdx.x;
  if (i >= total) return;
  int x  = i % wo;
  int y  = (i / wo) % ho;
  int bc = i / (wo * ho);
  const float* p = in + ((size_t)bc * hin + 2 * y) * win + 2 * x;
  out[i] = 0.25f * (p[0] + p[1] + p[win] + p[win + 1]);
}

// ---------------------------------------------------------------------------
// 3a. GroupNorm statistics: one block per (b, g)
// ---------------------------------------------------------------------------
__global__ void k_gn_stats(const float* __restrict__ xs, int N,
                           float* __restrict__ mu, float* __restrict__ rstd) {
  const int bg = blockIdx.x;
  const int b = bg / G_, g = bg % G_;
  const float* base = xs + ((size_t)b * C_ + g * CPG) * N;  // CPG*N contiguous
  const int M = CPG * N;
  float s = 0.f, ss = 0.f;
  for (int e = threadIdx.x; e < M; e += blockDim.x) {
    float v = base[e];
    s += v; ss += v * v;
  }
  __shared__ float sh0[256], sh1[256];
  sh0[threadIdx.x] = s; sh1[threadIdx.x] = ss;
  __syncthreads();
  for (int o = blockDim.x >> 1; o > 0; o >>= 1) {
    if ((int)threadIdx.x < o) { sh0[threadIdx.x] += sh0[threadIdx.x + o];
                                sh1[threadIdx.x] += sh1[threadIdx.x + o]; }
    __syncthreads();
  }
  if (threadIdx.x == 0) {
    float m   = sh0[0] / (float)M;
    float var = sh1[0] / (float)M - m * m;
    mu[bg]   = m;
    rstd[bg] = rsqrtf(var + EPS_);
  }
}

// ---------------------------------------------------------------------------
// 3b. Normalize + affine + transpose to [b][n][c] f16 for WMMA A-fragments
// ---------------------------------------------------------------------------
__global__ void k_norm(const float* __restrict__ xs,
                       const float* __restrict__ mu, const float* __restrict__ rstd,
                       const float* __restrict__ gnw, const float* __restrict__ gnb,
                       f16* __restrict__ xn, int N) {
  const int total = B_ * C_ * N;
  int i = blockIdx.x * blockDim.x + threadIdx.x;
  if (i >= total) return;
  int n = i % N;
  int c = (i / N) % C_;
  int b = i / (N * C_);
  int bg = b * G_ + c / CPG;
  float v = (xs[i] - mu[bg]) * rstd[bg] * gnw[c] + gnb[c];
  xn[((size_t)b * N + n) * C_ + c] = (f16)v;
}

// ---------------------------------------------------------------------------
// 4. QKV projection GEMM: [N x 128] @ [128 x 32]^T x3, one wave per 16 rows.
//    Q,K stored [b][n][i]; V stored TRANSPOSED [b][i][n] so that attention
//    B-fragments of V are contiguous reads.
// ---------------------------------------------------------------------------
__global__ void __launch_bounds__(128)
k_qkv(const f16* __restrict__ xn,
      const f16* __restrict__ qw, const f16* __restrict__ kw, const f16* __restrict__ vw,
      const float* __restrict__ qb, const float* __restrict__ kb, const float* __restrict__ vb,
      f16* __restrict__ q, f16* __restrict__ k, f16* __restrict__ vt, int N) {
  const int wave = threadIdx.x >> 5;
  const int tile = blockIdx.x * (blockDim.x >> 5) + wave;
  const int tilesPerB = N >> 4;
  const int b = tile / tilesPerB;
  const int t = tile % tilesPerB;
  const f16* xrow = xn + ((size_t)b * N + t * 16) * C_;

  v8f aq0 = {}, aq1 = {}, ak0 = {}, ak1 = {}, av0 = {}, av1 = {};
#pragma unroll
  for (int ks = 0; ks < 4; ++ks) {
    v16h a = load_a_rowmajor(xrow + ks * 32, C_);
    v16h bq0 = load_b_transposed(qw + 0 * C_  + ks * 32, C_);
    v16h bq1 = load_b_transposed(qw + 16 * C_ + ks * 32, C_);
    v16h bk0 = load_b_transposed(kw + 0 * C_  + ks * 32, C_);
    v16h bk1 = load_b_transposed(kw + 16 * C_ + ks * 32, C_);
    v16h bv0 = load_b_transposed(vw + 0 * C_  + ks * 32, C_);
    v16h bv1 = load_b_transposed(vw + 16 * C_ + ks * 32, C_);
    aq0 = wmma16(a, bq0, aq0);
    aq1 = wmma16(a, bq1, aq1);
    ak0 = wmma16(a, bk0, ak0);
    ak1 = wmma16(a, bk1, ak1);
    av0 = wmma16(a, bv0, av0);
    av1 = wmma16(a, bv1, av1);
  }
  const int lane = threadIdx.x & 31;
  const int n  = lane & 15;
  const int mh = (lane >> 4) << 3;
  const size_t ob = ((size_t)b * N + t * 16) * I_;
  const float bq0s = qb[n], bq1s = qb[16 + n];
  const float bk0s = kb[n], bk1s = kb[16 + n];
  const float bv0s = vb[n], bv1s = vb[16 + n];
  const size_t vt0 = ((size_t)b * I_ + n)      * N + t * 16 + mh;
  const size_t vt1 = ((size_t)b * I_ + 16 + n) * N + t * 16 + mh;
#pragma unroll
  for (int j = 0; j < 8; ++j) {
    const int m = j + mh;
    q[ob + m * I_ + n]      = (f16)(aq0[j] + bq0s);
    q[ob + m * I_ + 16 + n] = (f16)(aq1[j] + bq1s);
    k[ob + m * I_ + n]      = (f16)(ak0[j] + bk0s);
    k[ob + m * I_ + 16 + n] = (f16)(ak1[j] + bk1s);
    vt[vt0 + j] = (f16)(av0[j] + bv0s);   // contiguous 16B run per lane
    vt[vt1 + j] = (f16)(av1[j] + bv1s);
  }
}

// ---------------------------------------------------------------------------
// 5. Flash attention. Block = 4 waves = 4 consecutive 16-row Q tiles of one
//    batch. Per 64-key chunk: K[64x32] and Vt[32x64] are staged into LDS once
//    per block (async global->LDS when available), then each wave does
//    4 WMMAs for S, online softmax (lane-half shuffle reductions matching the
//    wave32 C/D layout), P staged D->A via per-wave LDS, 4 WMMAs for P.V.
// ---------------------------------------------------------------------------
__global__ void __launch_bounds__(128)
k_attn(const f16* __restrict__ q, const f16* __restrict__ k, const f16* __restrict__ vt,
       f16* __restrict__ att, int N, float inv_scale) {
  __shared__ __align__(16) f16 kbuf[64 * 32];      // keys row-major [key][i]
  __shared__ __align__(16) f16 vbuf[32 * 64];      // V transposed  [i][key]
  __shared__ __align__(16) f16 pbuf[4][16 * 64];   // per-wave P tile
  const int wave = threadIdx.x >> 5;
  const int tile = blockIdx.x * 4 + wave;
  const int tilesPerB = N >> 4;                    // multiple of 4 -> block-uniform b
  const int b = tile / tilesPerB;
  const int t = tile % tilesPerB;
  const f16* qbase  = q  + ((size_t)b * N + t * 16) * I_;
  const f16* kbase  = k  + (size_t)b * N * I_;
  const f16* vtbase = vt + (size_t)b * I_ * N;
  const int lane = threadIdx.x & 31;
  const int n  = lane & 15;
  const int mh = (lane >> 4) << 3;
  const int tid = threadIdx.x;
  f16* pb = pbuf[wave];

  v16h aq = load_a_rowmajor(qbase, I_);            // inter=32 == full WMMA K
  v8f acc0 = {}, acc1 = {};
  float rm[8], rl[8];
#pragma unroll
  for (int j = 0; j < 8; ++j) { rm[j] = -1e30f; rl[j] = 0.f; }

  for (int kb0 = 0; kb0 < N; kb0 += 64) {
    if (kb0 + 64 < N) {                            // gfx1250 global_prefetch_b8
      __builtin_prefetch(kbase + (size_t)(kb0 + 64) * I_, 0, 1);
      __builtin_prefetch(vtbase + kb0 + 64, 0, 1);
    }
    __syncthreads();                               // LDS reuse across chunks
    // stage K chunk: 64x32 f16 = 256 x 16B, 2 parts per thread
    {
      const f16* gk = kbase + (size_t)kb0 * I_;
#pragma unroll
      for (int r = 0; r < 2; ++r) {
        const int p = tid + r * 128;
        stage16(gk + p * 8, kbuf + p * 8);
      }
      // stage Vt chunk: 32 rows x 64 keys (row stride N in global)
#pragma unroll
      for (int r = 0; r < 2; ++r) {
        const int p = tid + r * 128;
        const int row = p >> 3, sub = p & 7;
        stage16(vtbase + (size_t)row * N + kb0 + sub * 8, vbuf + row * 64 + sub * 8);
      }
    }
#ifdef ASYNC_LDS
    __builtin_amdgcn_s_wait_asynccnt(0);           // drain this wave's async fills
#endif
    __syncthreads();

    // S = Q . K^T for 64 keys: 4 x 16-key tiles out of LDS
    v8f z = {};
    v8f s0 = wmma16(aq, load_b_transposed(kbuf +    0, 32), z);
    v8f s1 = wmma16(aq, load_b_transposed(kbuf +  512, 32), z);
    v8f s2 = wmma16(aq, load_b_transposed(kbuf + 1024, 32), z);
    v8f s3 = wmma16(aq, load_b_transposed(kbuf + 1536, 32), z);
#pragma unroll
    for (int j = 0; j < 8; ++j) {
      s0[j] *= inv_scale; s1[j] *= inv_scale;
      s2[j] *= inv_scale; s3[j] *= inv_scale;
    }

    // per-row chunk max across the 16-lane half
    float cm[8];
#pragma unroll
    for (int j = 0; j < 8; ++j)
      cm[j] = fmaxf(fmaxf(s0[j], s1[j]), fmaxf(s2[j], s3[j]));
#pragma unroll
    for (int msk = 1; msk < 16; msk <<= 1)
#pragma unroll
      for (int j = 0; j < 8; ++j) cm[j] = fmaxf(cm[j], __shfl_xor(cm[j], msk, 32));

    float alpha[8], cs[8];
#pragma unroll
    for (int j = 0; j < 8; ++j) {
      const float nm = fmaxf(rm[j], cm[j]);
      alpha[j] = __expf(rm[j] - nm);
      const float p0 = __expf(s0[j] - nm);
      const float p1 = __expf(s1[j] - nm);
      const float p2 = __expf(s2[j] - nm);
      const float p3 = __expf(s3[j] - nm);
      s0[j] = p0; s1[j] = p1; s2[j] = p2; s3[j] = p3;
      cs[j] = (p0 + p1) + (p2 + p3);
      rm[j] = nm;
    }
#pragma unroll
    for (int msk = 1; msk < 16; msk <<= 1)
#pragma unroll
      for (int j = 0; j < 8; ++j) cs[j] += __shfl_xor(cs[j], msk, 32);
#pragma unroll
    for (int j = 0; j < 8; ++j) {
      rl[j] = rl[j] * alpha[j] + cs[j];
      acc0[j] *= alpha[j];
      acc1[j] *= alpha[j];
    }

    // P: D-layout -> A-layout via per-wave LDS tile (16 rows x 64 keys)
#pragma unroll
    for (int j = 0; j < 8; ++j) {
      const int rbase = (j + mh) * 64;
      pb[rbase + n]      = (f16)s0[j];
      pb[rbase + 16 + n] = (f16)s1[j];
      pb[rbase + 32 + n] = (f16)s2[j];
      pb[rbase + 48 + n] = (f16)s3[j];
    }
    v16h ap0 = load_a_rowmajor(pb, 64);        // keys 0-31 of chunk
    v16h ap1 = load_a_rowmajor(pb + 32, 64);   // keys 32-63 of chunk
    // B = V chunk from transposed LDS tile: B[k][n] = vbuf[(i0+n)*64 + k0+k]
    acc0 = wmma16(ap0, load_b_transposed(vbuf +  0 * 64 +  0, 64), acc0);
    acc1 = wmma16(ap0, load_b_transposed(vbuf + 16 * 64 +  0, 64), acc1);
    acc0 = wmma16(ap1, load_b_transposed(vbuf +  0 * 64 + 32, 64), acc0);
    acc1 = wmma16(ap1, load_b_transposed(vbuf + 16 * 64 + 32, 64), acc1);
  }

  const size_t ob = ((size_t)b * N + t * 16) * I_;
#pragma unroll
  for (int j = 0; j < 8; ++j) {
    const float inv = 1.0f / rl[j];
    att[ob + (j + mh) * I_ + n]      = (f16)(acc0[j] * inv);
    att[ob + (j + mh) * I_ + 16 + n] = (f16)(acc1[j] * inv);
  }
}

// ---------------------------------------------------------------------------
// 6. O-projection [N x 32] @ [32 x 128]^T + bias, gamma*o + residual.
// ---------------------------------------------------------------------------
__global__ void __launch_bounds__(128)
k_oproj(const f16* __restrict__ att, const f16* __restrict__ ow,
        const float* __restrict__ obias, const float* __restrict__ gamma_p, int iscale,
        const float* __restrict__ xs, float* __restrict__ y, int N) {
  const int wave = threadIdx.x >> 5;
  const int tile = blockIdx.x * (blockDim.x >> 5) + wave;
  const int tilesPerB = N >> 4;
  const int b = tile / tilesPerB;
  const int t = tile % tilesPerB;
  v16h a = load_a_rowmajor(att + ((size_t)b * N + t * 16) * I_, I_);
  const int lane = threadIdx.x & 31;
  const int n  = lane & 15;
  const int mh = (lane >> 4) << 3;
  const float gamma = gamma_p[iscale];
#pragma unroll
  for (int ct = 0; ct < 8; ++ct) {
    v16h bw = load_b_transposed(ow + (size_t)ct * 16 * I_, I_);
    v8f z = {};
    v8f o = wmma16(a, bw, z);
    const int c = ct * 16 + n;
    const float bias = obias[c];
#pragma unroll
    for (int j = 0; j < 8; ++j) {
      const int pix = t * 16 + j + mh;
      const size_t idx = ((size_t)b * C_ + c) * N + pix;
      y[idx] = gamma * (o[j] + bias) + xs[idx];
    }
  }
}

// ---------------------------------------------------------------------------
// 7. Bilinear upsample (reference coords) + softmax-weighted fusion
// ---------------------------------------------------------------------------
__device__ __forceinline__ float bilerp(const float* __restrict__ p, int hw, int oy, int ox) {
  const float r = (float)hw / (float)H_;
  float cy = (oy + 0.5f) * r - 0.5f;
  cy = fminf(fmaxf(cy, 0.f), (float)hw - 1.f);
  int y0 = (int)floorf(cy); int y1 = y0 + 1 < hw ? y0 + 1 : hw - 1;
  float ty = cy - (float)y0;
  float cx = (ox + 0.5f) * r - 0.5f;
  cx = fminf(fmaxf(cx, 0.f), (float)hw - 1.f);
  int x0 = (int)floorf(cx); int x1 = x0 + 1 < hw ? x0 + 1 : hw - 1;
  float tx = cx - (float)x0;
  float v00 = p[y0 * hw + x0], v01 = p[y0 * hw + x1];
  float v10 = p[y1 * hw + x0], v11 = p[y1 * hw + x1];
  float top = v00 * (1.f - tx) + v01 * tx;
  float bot = v10 * (1.f - tx) + v11 * tx;
  return top * (1.f - ty) + bot * ty;
}

__global__ void k_fuse(const float* __restrict__ y1, const float* __restrict__ y2,
                       const float* __restrict__ y4, const float* __restrict__ sw,
                       float* __restrict__ out) {
  const int total = B_ * C_ * H_ * W_;
  int i = blockIdx.x * blockDim.x + threadIdx.x;
  if (i >= total) return;
  float w0 = sw[0], w1 = sw[1], w2 = sw[2];
  float mx = fmaxf(w0, fmaxf(w1, w2));
  float e0 = __expf(w0 - mx), e1 = __expf(w1 - mx), e2 = __expf(w2 - mx);
  float inv = 1.f / (e0 + e1 + e2);
  e0 *= inv; e1 *= inv; e2 *= inv;
  int x  = i % W_;
  int yy = (i / W_) % H_;
  int bc = i / (H_ * W_);
  float f1 = y1[i];
  float f2 = bilerp(y2 + (size_t)bc * 32 * 32, 32, yy, x);
  float f4 = bilerp(y4 + (size_t)bc * 16 * 16, 16, yy, x);
  out[i] = e0 * f1 + e1 * f2 + e2 * f4;
}

// ---------------------------------------------------------------------------
// Host launcher
// ---------------------------------------------------------------------------
extern "C" void kernel_launch(void* const* d_in, const int* in_sizes, int n_in,
                              void* d_out, int out_size, void* d_ws, size_t ws_size,
                              hipStream_t stream) {
  const float* x     = (const float*)d_in[0];
  const float* gn_w  = (const float*)d_in[1];
  const float* gn_b  = (const float*)d_in[2];
  const float* qw    = (const float*)d_in[3];
  const float* qb    = (const float*)d_in[4];
  const float* kw    = (const float*)d_in[5];
  const float* kb    = (const float*)d_in[6];
  const float* vw    = (const float*)d_in[7];
  const float* vb    = (const float*)d_in[8];
  const float* ow    = (const float*)d_in[9];
  const float* ob    = (const float*)d_in[10];
  const float* gamma = (const float*)d_in[11];
  const float* sw    = (const float*)d_in[12];
  float* out = (float*)d_out;

  const int Ns[S_] = {H_ * W_, (H_ / 2) * (W_ / 2), (H_ / 4) * (W_ / 4)};  // 4096,1024,256

  char* wsp = (char*)d_ws;
  size_t off = 0;
  auto alloc = [&](size_t bytes) -> void* {
    void* p = wsp + off;
    off = (off + bytes + 255) & ~(size_t)255;
    return p;
  };

  float* xs2 = (float*)alloc(sizeof(float) * B_ * C_ * Ns[1]);
  float* xs4 = (float*)alloc(sizeof(float) * B_ * C_ * Ns[2]);
  f16* qwh = (f16*)alloc(sizeof(f16) * S_ * I_ * C_);
  f16* kwh = (f16*)alloc(sizeof(f16) * S_ * I_ * C_);
  f16* vwh = (f16*)alloc(sizeof(f16) * S_ * I_ * C_);
  f16* owh = (f16*)alloc(sizeof(f16) * S_ * C_ * I_);
  f16*   xn[S_]; f16* qm[S_]; f16* km[S_]; f16* vm[S_]; f16* am[S_];
  float* ym[S_]; float* mu[S_]; float* rs[S_];
  for (int i = 0; i < S_; ++i) {
    xn[i] = (f16*)alloc(sizeof(f16) * B_ * Ns[i] * C_);
    qm[i] = (f16*)alloc(sizeof(f16) * B_ * Ns[i] * I_);
    km[i] = (f16*)alloc(sizeof(f16) * B_ * Ns[i] * I_);
    vm[i] = (f16*)alloc(sizeof(f16) * B_ * Ns[i] * I_);   // V transposed [b][i][n]
    am[i] = (f16*)alloc(sizeof(f16) * B_ * Ns[i] * I_);
    ym[i] = (float*)alloc(sizeof(float) * B_ * C_ * Ns[i]);
    mu[i] = (float*)alloc(sizeof(float) * B_ * G_);
    rs[i] = (float*)alloc(sizeof(float) * B_ * G_);
  }

  // 1. weight conversion
  {
    const int nw = S_ * I_ * C_;
    const int blk = (nw + 255) / 256;
    k_f32_to_f16<<<blk, 256, 0, stream>>>(qw, qwh, nw);
    k_f32_to_f16<<<blk, 256, 0, stream>>>(kw, kwh, nw);
    k_f32_to_f16<<<blk, 256, 0, stream>>>(vw, vwh, nw);
    k_f32_to_f16<<<blk, 256, 0, stream>>>(ow, owh, nw);
  }

  // 2. downsample chain
  k_down2<<<(B_ * C_ * Ns[1] + 255) / 256, 256, 0, stream>>>(x, xs2, H_, W_);
  k_down2<<<(B_ * C_ * Ns[2] + 255) / 256, 256, 0, stream>>>(xs2, xs4, H_ / 2, W_ / 2);

  const float* xsp[S_] = {x, xs2, xs4};
  const float inv_scale = 1.0f / sqrtf((float)I_);  // TEMPERATURE = 1

  for (int i = 0; i < S_; ++i) {
    const int N = Ns[i];
    // 3. group norm
    k_gn_stats<<<B_ * G_, 256, 0, stream>>>(xsp[i], N, mu[i], rs[i]);
    k_norm<<<(B_ * C_ * N + 255) / 256, 256, 0, stream>>>(
        xsp[i], mu[i], rs[i], gn_w + i * C_, gn_b + i * C_, xn[i], N);
    // 4..6: WMMA pipeline, one wave per 16-row tile, 4 waves per block
    const int tiles  = B_ * (N >> 4);
    const int blocks = tiles / 4;
    k_qkv<<<blocks, 128, 0, stream>>>(
        xn[i], qwh + i * I_ * C_, kwh + i * I_ * C_, vwh + i * I_ * C_,
        qb + i * I_, kb + i * I_, vb + i * I_, qm[i], km[i], vm[i], N);
    k_attn<<<blocks, 128, 0, stream>>>(qm[i], km[i], vm[i], am[i], N, inv_scale);
    k_oproj<<<blocks, 128, 0, stream>>>(am[i], owh + i * C_ * I_, ob + i * C_,
                                        gamma, i, xsp[i], ym[i], N);
  }

  // 7. fuse
  k_fuse<<<(B_ * C_ * H_ * W_ + 255) / 256, 256, 0, stream>>>(ym[0], ym[1], ym[2], sw, out);
}
